// MLNet3_31284541784583
// MI455X (gfx1250) — compile-verified, matrix operand order checked by reading, not run on hardware
//
#include <hip/hip_runtime.h>

// ---------------------------------------------------------------------------
// MLNet3 GraphNet forward for MI455X (gfx1250).
// Dense math: v_wmma_f32_16x16x32_f16 (f32 -> f16 operands, f32 accumulate).
// Gather+concat MLP inputs fused into the GEMM A-loader (templated on #segs).
// B tiles staged cooperatively in LDS (ds_store/ds_load + split barrier),
// double-buffered. Scatter/segment ops use global_atomic_add_f32.
// ---------------------------------------------------------------------------

typedef __attribute__((ext_vector_type(16))) _Float16 v16h;
typedef __attribute__((ext_vector_type(8)))  _Float16 v8h;
typedef __attribute__((ext_vector_type(8)))  float    v8f;

constexpr int Nn = 32768;
constexpr int Ee = 131072;
constexpr int Bg = 1024;
constexpr int H  = 128;
constexpr int NF = 40;
constexpr int EF = 10;
constexpr int GF = 200;
constexpr int Tt = 12;
constexpr int DEPTH = 3;
constexpr float BN_EPS = 1e-5f;

// A-operand: up to 4 concatenated segments along K, each with an optional
// row-gather index (covers concat([x[row], x[col], e, u[b_row]]) etc).
struct ASegs {
  const float* ptr[4];
  const int*   idx[4];
  int kstart[5];   // prefix offsets in K; kstart[nseg] == K
  int ld[4];       // row stride in elements
  int nseg;
};

// ---------------------------------------------------------------------------
// Fragment helpers. CDNA5 16-bit A/B 16x32 lane layout: lanes 0-15 kbase=0,
// lanes 16-31 kbase=8; elements 0..7 <- k=kbase+0..7, 8..15 <- k=kbase+16..23.
// ---------------------------------------------------------------------------
__device__ __forceinline__ void cvt8(_Float16* dst, const float* p) {
  const float2* p2 = (const float2*)p;     // all row strides even -> 8B aligned
#pragma unroll
  for (int j = 0; j < 4; ++j) {
    float2 t = p2[j];
    dst[2 * j]     = (_Float16)t.x;
    dst[2 * j + 1] = (_Float16)t.y;
  }
}

// Guarded variant for the (wave-uniform) tail chunk of a segment.
__device__ __forceinline__ void cvt8_tail(_Float16* dst, const float* p, int valid) {
#pragma unroll
  for (int j = 0; j < 8; ++j)
    dst[j] = (j < valid) ? (_Float16)p[j] : (_Float16)0.0f;
}

// ---------------------------------------------------------------------------
// WMMA GEMM: out[M,O] = act(A[M,K] @ W[O,K]^T + bias)
// Block = 128 threads = 4 waves; all waves share one 64-wide N strip (nt) and
// take 4 consecutive 16-row M tiles. B chunk (64 cols x 32 k, f16) staged in
// LDS once per block per chunk; each wave runs 4 WMMAs per chunk.
// ---------------------------------------------------------------------------
template <int NSEG>
__global__ __launch_bounds__(128)
void gemm_wmma_kernel(ASegs S, const float* __restrict__ W, const float* __restrict__ bias,
                      float* __restrict__ out, int M, int K, int O, int relu) {
  __shared__ v8h ldsB[2][4][32][2];        // [buf][sub][lane][half] = 8 KB

  const int tid  = threadIdx.x;
  const int lane = tid & 31;
  const int wave = tid >> 5;

  const int ntiles  = (O + 63) >> 6;
  const int mtiles  = (M + 15) >> 4;
  const int nt = blockIdx.x % ntiles;
  const int mt = (blockIdx.x / ntiles) * 4 + wave;   // all calls have mtiles % 4 == 0

  // ---- A row bases (gather resolved once; clamp => safe addresses, rows we
  //      never store may hold garbage, which is fine) ----
  const int mlocal = lane & 15;
  const int kb = (lane >> 4) << 3;                   // 0 or 8
  int m = mt * 16 + mlocal;
  if (m >= M) m = 0;
  const float* rowbase[NSEG];
#pragma unroll
  for (int s = 0; s < NSEG; ++s) {
    int r = S.idx[s] ? S.idx[s][m] : m;
    rowbase[s] = S.ptr[s] + (size_t)r * S.ld[s];
  }

  // ---- B loader role: this thread owns fragment (sub = wave, lane) ----
  int nload = nt * 64 + wave * 16 + mlocal;
  if (nload >= O) nload = 0;                         // safe address, unused column
  const float* wbase = W + (size_t)nload * K;

  v8f acc[4] = {};
  int buf = 0;

#pragma unroll
  for (int s = 0; s < NSEG; ++s) {
    const int ks   = S.kstart[s];
    const int segK = S.kstart[s + 1] - ks;
    const float* abase = rowbase[s];
    const float* wseg  = wbase + ks;

    for (int c = 0; c < segK; c += 32) {             // uniform chunk loop
      const bool full = (c + 32 <= segK);            // wave-uniform (scalar branch)

      // ---- stage my B fragment into LDS ----
      {
        _Float16 tb[16];
        if (full) {
          cvt8(tb,     wseg + c + kb);
          cvt8(tb + 8, wseg + c + kb + 16);
        } else {
          cvt8_tail(tb,     wseg + c + kb,      segK - (c + kb));
          cvt8_tail(tb + 8, wseg + c + kb + 16, segK - (c + kb + 16));
        }
        v8h lo, hi;
#pragma unroll
        for (int j = 0; j < 8; ++j) { lo[j] = tb[j]; hi[j] = tb[8 + j]; }
        ldsB[buf][wave][lane][0] = lo;               // ds_store_b128
        ldsB[buf][wave][lane][1] = hi;
      }

      // ---- A fragment (global, per wave) ----
      __builtin_prefetch(abase + c + 32);            // global_prefetch_b8
      _Float16 ta[16];
      if (full) {
        cvt8(ta,     abase + c + kb);
        cvt8(ta + 8, abase + c + kb + 16);
      } else {
        cvt8_tail(ta,     abase + c + kb,      segK - (c + kb));
        cvt8_tail(ta + 8, abase + c + kb + 16, segK - (c + kb + 16));
      }
      v16h afrag;
#pragma unroll
      for (int j = 0; j < 16; ++j) afrag[j] = ta[j];

      __syncthreads();                               // B chunk visible to all waves

      // ---- 4 WMMAs on the shared B chunk ----
#pragma unroll
      for (int sub = 0; sub < 4; ++sub) {
        v8h lo = ldsB[buf][sub][lane][0];            // ds_load_b128
        v8h hi = ldsB[buf][sub][lane][1];
        v16h bfrag;
#pragma unroll
        for (int j = 0; j < 8; ++j) { bfrag[j] = lo[j]; bfrag[8 + j] = hi[j]; }
        acc[sub] = __builtin_amdgcn_wmma_f32_16x16x32_f16(
            false, afrag, false, bfrag, (short)0, acc[sub], false, false);
      }
      buf ^= 1;                                      // double buffer
    }
  }

  // ---- epilogue: C/D layout (VGPR v -> M = v or 8+v; col = lane&15) ----
  if (mt < mtiles) {
    const int mbase = mt * 16 + ((lane >> 4) << 3);
#pragma unroll
    for (int sub = 0; sub < 4; ++sub) {
      int n = nt * 64 + sub * 16 + mlocal;
      if (n >= O) continue;
      float bv = bias ? bias[n] : 0.0f;
#pragma unroll
      for (int v = 0; v < 8; ++v) {
        int mrow = mbase + v;
        if (mrow < M) {
          float val = acc[sub][v] + bv;
          if (relu) val = fmaxf(val, 0.0f);
          out[(size_t)mrow * O + n] = val;
        }
      }
    }
  }
}

// ---------------------------------------------------------------------------
// Elementwise / scatter kernels (grid-stride, blockDim = 256)
// ---------------------------------------------------------------------------
__global__ void k_zero(float* __restrict__ p, long n) {
  for (long i = (long)blockIdx.x * 256 + threadIdx.x; i < n; i += (long)gridDim.x * 256)
    p[i] = 0.0f;
}

__global__ void k_brow(const int* __restrict__ row, const int* __restrict__ batch,
                       int* __restrict__ brow, int E) {
  for (long i = (long)blockIdx.x * 256 + threadIdx.x; i < E; i += (long)gridDim.x * 256)
    brow[i] = batch[row[i]];
}

__global__ void k_count(const int* __restrict__ seg, float* __restrict__ cnt, long n) {
  for (long i = (long)blockIdx.x * 256 + threadIdx.x; i < n; i += (long)gridDim.x * 256)
    atomicAdd(&cnt[seg[i]], 1.0f);
}

__global__ void k_scatter_add(const float* __restrict__ vals, const int* __restrict__ seg,
                              float* __restrict__ out, int F, long n) {
  for (long i = (long)blockIdx.x * 256 + threadIdx.x; i < n; i += (long)gridDim.x * 256) {
    long e = i / F;
    int  f = (int)(i - e * F);
    atomicAdd(&out[(long)seg[e] * F + f], vals[i]);
  }
}

__global__ void k_div_rows(float* __restrict__ out, const float* __restrict__ cnt,
                           int F, long n) {
  for (long i = (long)blockIdx.x * 256 + threadIdx.x; i < n; i += (long)gridDim.x * 256) {
    long b = i / F;
    out[i] /= fmaxf(cnt[b], 1.0f);
  }
}

// fused size_norm (optional) + eval-mode BatchNorm1d
__global__ void k_norm_bn(float* __restrict__ x, const int* __restrict__ seg,
                          const float* __restrict__ deg,
                          const float* __restrict__ g, const float* __restrict__ bb,
                          const float* __restrict__ mm, const float* __restrict__ vv,
                          int F, long n) {
  for (long i = (long)blockIdx.x * 256 + threadIdx.x; i < n; i += (long)gridDim.x * 256) {
    long row = i / F;
    int  f = (int)(i - row * F);
    float val = x[i];
    if (deg) val *= rsqrtf(fmaxf(deg[seg[row]], 1.0f));
    val = (val - mm[f]) * (g[f] * rsqrtf(vv[f] + BN_EPS)) + bb[f];
    x[i] = val;
  }
}

// agg[col[e], 0:128] += x[row[e]] ; agg[col[e], 128:256] += e_feat[e]
__global__ void k_agg(const float* __restrict__ x, const float* __restrict__ ef,
                      const int* __restrict__ row, const int* __restrict__ col,
                      float* __restrict__ agg, int E) {
  long n = (long)E * H;
  for (long i = (long)blockIdx.x * 256 + threadIdx.x; i < n; i += (long)gridDim.x * 256) {
    long ed = i >> 7;
    int  f  = (int)(i & 127);
    int  c  = col[ed];
    int  r  = row[ed];
    atomicAdd(&agg[(long)c * 256 + f],       x[(long)r * 128 + f]);
    atomicAdd(&agg[(long)c * 256 + 128 + f], ef[i]);
  }
}

// ---------------------------------------------------------------------------
// Host side
// ---------------------------------------------------------------------------
static inline int eltGrid(long n) {
  long g = (n + 255) / 256;
  if (g > 1048576) g = 1048576;
  if (g < 1) g = 1;
  return (int)g;
}

static inline ASegs seg1(const float* p, int ld, int K, const int* idx = nullptr) {
  ASegs s{};
  s.ptr[0] = p; s.idx[0] = idx; s.ld[0] = ld;
  s.kstart[0] = 0; s.kstart[1] = K; s.nseg = 1;
  return s;
}

static void gemm(hipStream_t st, const ASegs& A, const float* W, const float* bias,
                 float* out, int M, int K, int O, int relu) {
  int mtiles4 = (M + 63) / 64;
  int ntiles  = (O + 63) / 64;
  dim3 grid(mtiles4 * ntiles), block(128);
  switch (A.nseg) {
    case 1: gemm_wmma_kernel<1><<<grid, block, 0, st>>>(A, W, bias, out, M, K, O, relu); break;
    case 2: gemm_wmma_kernel<2><<<grid, block, 0, st>>>(A, W, bias, out, M, K, O, relu); break;
    case 3: gemm_wmma_kernel<3><<<grid, block, 0, st>>>(A, W, bias, out, M, K, O, relu); break;
    default: gemm_wmma_kernel<4><<<grid, block, 0, st>>>(A, W, bias, out, M, K, O, relu); break;
  }
}

extern "C" void kernel_launch(void* const* d_in, const int* in_sizes, int n_in,
                              void* d_out, int out_size, void* d_ws, size_t ws_size,
                              hipStream_t stream) {
  (void)in_sizes; (void)n_in; (void)out_size; (void)ws_size;

  // --- inputs (setup_inputs dict order) ---
  const float* x_in  = (const float*)d_in[0];
  const int*   ei    = (const int*)d_in[1];
  const float* eattr = (const float*)d_in[2];
  const int*   batch = (const int*)d_in[3];
  const float* gattr = (const float*)d_in[4];
  const int* row = ei;
  const int* col = ei + Ee;

  auto P = [&](int i) { return (const float*)d_in[i]; };
  // params flattened depth-first in insertion order, 4 tensors per MLP (w1,b1,w2,b2)
  const int MLP1  = 17;
  const int GN0   = 21;   // gn layer i at GN0+16*i: edge(+0) node1(+4) node2(+8) glob(+12)
  const int BN_N  = 69, BN_E = 85, BN_G = 101;  // bn j at base+4*j: g,b,m,v
  auto bn = [&](int base, int j, int comp) { return P(base + 4 * j + comp); };

  // --- workspace carve ---
  float* wsf = (float*)d_ws;
  auto take = [&](long n) { float* r = wsf; wsf += n; return r; };
  float* xa    = take((long)Nn * H);
  float* xb    = take((long)Nn * H);
  float* ea    = take((long)Ee * H);
  float* eb    = take((long)Ee * H);
  float* tmpE  = take((long)Ee * H);   // hidden-layer scratch (largest M)
  float* ua    = take((long)Bg * H);
  float* ub    = take((long)Bg * H);
  float* agg   = take((long)Nn * 2 * H);
  float* agg2  = take((long)Nn * H);
  float* ninfo = take((long)Bg * H);
  float* einfo = take((long)Bg * H);
  float* hidB  = take((long)Bg * H);
  float* u0    = take((long)Bg * EF);
  float* degN  = take(Bg);
  float* degE  = take(Bg);
  int*   brow  = (int*)take(Ee);

  auto zero = [&](float* p, long n) { k_zero<<<eltGrid(n), 256, 0, stream>>>(p, n); };

  // --- prologue: degrees, b_row, u0 = scatter_mean(edge_attr, b_row) ---
  zero(degN, Bg);
  zero(degE, Bg);
  zero(u0, (long)Bg * EF);
  k_brow<<<eltGrid(Ee), 256, 0, stream>>>(row, batch, brow, Ee);
  k_count<<<eltGrid(Nn), 256, 0, stream>>>(batch, degN, Nn);
  k_count<<<eltGrid(Ee), 256, 0, stream>>>(brow, degE, Ee);
  k_scatter_add<<<eltGrid((long)Ee * EF), 256, 0, stream>>>(eattr, brow, u0, EF, (long)Ee * EF);
  k_div_rows<<<eltGrid((long)Bg * EF), 256, 0, stream>>>(u0, degE, EF, (long)Bg * EF);

  // encoders: x = mlp_node(x_in), e = mlp_edge(edge_attr), u = mlp_global(u0)
  gemm(stream, seg1(x_in, NF, NF), P(5), P(6), tmpE, Nn, NF, H, 1);
  gemm(stream, seg1(tmpE, H, H),   P(7), P(8), xa,   Nn, H,  H, 0);
  gemm(stream, seg1(eattr, EF, EF), P(9),  P(10), tmpE, Ee, EF, H, 1);
  gemm(stream, seg1(tmpE, H, H),    P(11), P(12), ea,   Ee, H,  H, 0);
  gemm(stream, seg1(u0, EF, EF), P(13), P(14), hidB, Bg, EF, H, 1);
  gemm(stream, seg1(hidB, H, H), P(15), P(16), ua,   Bg, H,  H, 0);

  // size_norm + bn (last BN index = DEPTH)
  k_norm_bn<<<eltGrid((long)Nn * H), 256, 0, stream>>>(xa, batch, degN,
      bn(BN_N, DEPTH, 0), bn(BN_N, DEPTH, 1), bn(BN_N, DEPTH, 2), bn(BN_N, DEPTH, 3), H, (long)Nn * H);
  k_norm_bn<<<eltGrid((long)Ee * H), 256, 0, stream>>>(ea, brow, degE,
      bn(BN_E, DEPTH, 0), bn(BN_E, DEPTH, 1), bn(BN_E, DEPTH, 2), bn(BN_E, DEPTH, 3), H, (long)Ee * H);
  k_norm_bn<<<eltGrid((long)Bg * H), 256, 0, stream>>>(ua, nullptr, nullptr,
      bn(BN_G, DEPTH, 0), bn(BN_G, DEPTH, 1), bn(BN_G, DEPTH, 2), bn(BN_G, DEPTH, 3), H, (long)Bg * H);

  // --- GN layers ---
  float* xc = xa; float* xn = xb;
  float* ec = ea; float* en = eb;
  float* uc = ua; float* un = ub;
  for (int i = 0; i < DEPTH; ++i) {
    int gb = GN0 + 16 * i;

    // edge model: e = mlp(concat[x[row], x[col], e, u[b_row]])   (gather fused into GEMM)
    ASegs sE{};
    sE.nseg = 4;
    sE.ptr[0] = xc; sE.idx[0] = row;     sE.ld[0] = H;
    sE.ptr[1] = xc; sE.idx[1] = col;     sE.ld[1] = H;
    sE.ptr[2] = ec; sE.idx[2] = nullptr; sE.ld[2] = H;
    sE.ptr[3] = uc; sE.idx[3] = brow;    sE.ld[3] = H;
    sE.kstart[0] = 0; sE.kstart[1] = 128; sE.kstart[2] = 256; sE.kstart[3] = 384; sE.kstart[4] = 512;
    gemm(stream, sE, P(gb + 0), P(gb + 1), tmpE, Ee, 4 * H, H, 1);
    gemm(stream, seg1(tmpE, H, H), P(gb + 2), P(gb + 3), en, Ee, H, H, 0);

    // node model: agg = segment_sum(concat[x[row], e], col); mlp; concat; mlp
    zero(agg, (long)Nn * 2 * H);
    k_agg<<<eltGrid((long)Ee * H), 256, 0, stream>>>(xc, en, row, col, agg, Ee);
    gemm(stream, seg1(agg, 2 * H, 2 * H), P(gb + 4), P(gb + 5), tmpE, Nn, 2 * H, H, 1);
    gemm(stream, seg1(tmpE, H, H),        P(gb + 6), P(gb + 7), agg2, Nn, H, H, 0);
    ASegs sN{};
    sN.nseg = 3;
    sN.ptr[0] = xc;   sN.idx[0] = nullptr; sN.ld[0] = H;
    sN.ptr[1] = agg2; sN.idx[1] = nullptr; sN.ld[1] = H;
    sN.ptr[2] = uc;   sN.idx[2] = batch;   sN.ld[2] = H;
    sN.kstart[0] = 0; sN.kstart[1] = 128; sN.kstart[2] = 256; sN.kstart[3] = 384;
    gemm(stream, sN, P(gb + 8), P(gb + 9), tmpE, Nn, 3 * H, H, 1);
    gemm(stream, seg1(tmpE, H, H), P(gb + 10), P(gb + 11), xn, Nn, H, H, 0);

    // global model: u = mlp(concat[u, scatter_mean(x), scatter_mean(e)])
    zero(einfo, (long)Bg * H);
    zero(ninfo, (long)Bg * H);
    k_scatter_add<<<eltGrid((long)Ee * H), 256, 0, stream>>>(en, brow, einfo, H, (long)Ee * H);
    k_div_rows<<<eltGrid((long)Bg * H), 256, 0, stream>>>(einfo, degE, H, (long)Bg * H);
    k_scatter_add<<<eltGrid((long)Nn * H), 256, 0, stream>>>(xn, batch, ninfo, H, (long)Nn * H);
    k_div_rows<<<eltGrid((long)Bg * H), 256, 0, stream>>>(ninfo, degN, H, (long)Bg * H);
    ASegs sG{};
    sG.nseg = 3;
    sG.ptr[0] = uc;    sG.idx[0] = nullptr; sG.ld[0] = H;
    sG.ptr[1] = ninfo; sG.idx[1] = nullptr; sG.ld[1] = H;
    sG.ptr[2] = einfo; sG.idx[2] = nullptr; sG.ld[2] = H;
    sG.kstart[0] = 0; sG.kstart[1] = 128; sG.kstart[2] = 256; sG.kstart[3] = 384;
    gemm(stream, sG, P(gb + 12), P(gb + 13), hidB, Bg, 3 * H, H, 1);
    gemm(stream, seg1(hidB, H, H), P(gb + 14), P(gb + 15), un, Bg, H, H, 0);

    // size_norm + bn (index i)
    k_norm_bn<<<eltGrid((long)Nn * H), 256, 0, stream>>>(xn, batch, degN,
        bn(BN_N, i, 0), bn(BN_N, i, 1), bn(BN_N, i, 2), bn(BN_N, i, 3), H, (long)Nn * H);
    k_norm_bn<<<eltGrid((long)Ee * H), 256, 0, stream>>>(en, brow, degE,
        bn(BN_E, i, 0), bn(BN_E, i, 1), bn(BN_E, i, 2), bn(BN_E, i, 3), H, (long)Ee * H);
    k_norm_bn<<<eltGrid((long)Bg * H), 256, 0, stream>>>(un, nullptr, nullptr,
        bn(BN_G, i, 0), bn(BN_G, i, 1), bn(BN_G, i, 2), bn(BN_G, i, 3), H, (long)Bg * H);

    float* t;
    t = xc; xc = xn; xn = t;
    t = ec; ec = en; en = t;
    t = uc; uc = un; un = t;
  }

  // --- readout: out = mlp1(concat[u, graph_attr]) ---
  ASegs sF{};
  sF.nseg = 2;
  sF.ptr[0] = uc;    sF.idx[0] = nullptr; sF.ld[0] = H;
  sF.ptr[1] = gattr; sF.idx[1] = nullptr; sF.ld[1] = GF;
  sF.kstart[0] = 0; sF.kstart[1] = H; sF.kstart[2] = H + GF;
  gemm(stream, sF, P(MLP1 + 0), P(MLP1 + 1), hidB, Bg, H + GF, H, 1);
  gemm(stream, seg1(hidB, H, H), P(MLP1 + 2), P(MLP1 + 3), (float*)d_out, Bg, H, Tt, 0);
}
